// HungarianMatcher_79757542686916
// MI455X (gfx1250) — compile-verified, matrix operand order checked by reading, not run on hardware
//
#include <hip/hip_runtime.h>
#include <hip/hip_bf16.h>

// ---------------------------------------------------------------------------
// HungarianMatcher cost matrix for MI455X (gfx1250, wave32, WMMA).
//
// Shapes: B=4 Q=100 C1=81 H=W=256 N=50 P=12544  -> out [4,100,200] f32.
// Math:
//   cost_mask = ((-os) @ ts^T + rowsum(softplus(os))) / P        (1 WMMA GEMM)
//   cost_dice = 1 - (2*(sig(os) @ ts^T) + 1) / (rowsum(sig)+colsum(ts)+1)
//   cost_cls  = -softmax(logits)[r, tgt_ids[c]]
//   C = 5*cost_mask + 2*cost_cls + 5*cost_dice, nan_to_num
// GEMMs: M=400, N=200 (pad 208), K=12544 = 392 * 32, f16 in / f32 accum via
// v_wmma_f32_16x16x32_f16. Operands ~25MB -> resident in 192MB L2.
// ---------------------------------------------------------------------------

typedef __attribute__((ext_vector_type(16))) _Float16 v16h;
typedef __attribute__((ext_vector_type(8)))  _Float16 v8h;
typedef __attribute__((ext_vector_type(8)))  float    v8f;

#define NPTS  12544
#define HH    256
#define WW    256
#define NROW  400     // B*Q
#define NCOL  200     // B*N
#define NCOLP 208     // padded to 13*16
#define NCLS  81
#define MT    25      // 400/16
#define NT    13      // 208/16
#define HW    (HH*WW)

struct Tap { int o[4]; float w[4]; };   // 32B: 4 gather offsets + 4 weights

// ---------------- Kernel 1: precompute bilinear taps (shared by all masks) --
__global__ void __launch_bounds__(256)
k_taps(const float* __restrict__ coords, Tap* __restrict__ taps) {
    int p = blockIdx.x * 256 + threadIdx.x;
    if (p >= NPTS) return;
    float cx = coords[2*p+0], cy = coords[2*p+1];
    float x = cx * (float)WW - 0.5f;
    float y = cy * (float)HH - 0.5f;
    float x0f = floorf(x), y0f = floorf(y);
    int ix0 = (int)x0f, iy0 = (int)y0f;
    int ix1 = ix0 + 1,  iy1 = iy0 + 1;
    float wx1 = x - x0f, wx0 = 1.f - wx1;
    float wy1 = y - y0f, wy0 = 1.f - wy1;
    bool vx0 = (ix0 >= 0) && (ix0 < WW), vx1 = (ix1 >= 0) && (ix1 < WW);
    bool vy0 = (iy0 >= 0) && (iy0 < HH), vy1 = (iy1 >= 0) && (iy1 < HH);
    int cx0 = min(max(ix0, 0), WW-1), cx1 = min(max(ix1, 0), WW-1);
    int cy0 = min(max(iy0, 0), HH-1), cy1 = min(max(iy1, 0), HH-1);
    Tap t;
    t.o[0] = cy0*WW + cx0;  t.w[0] = (vy0 && vx0) ? wy0*wx0 : 0.f;
    t.o[1] = cy0*WW + cx1;  t.w[1] = (vy0 && vx1) ? wy0*wx1 : 0.f;
    t.o[2] = cy1*WW + cx0;  t.w[2] = (vy1 && vx0) ? wy1*wx0 : 0.f;
    t.o[3] = cy1*WW + cx1;  t.w[3] = (vy1 && vx1) ? wy1*wx1 : 0.f;
    taps[p] = t;
}

// ---- Kernel 2: sample pred masks; emit A1=-os, A2=sigmoid(os) (f16) + sums -
// One block per row => fixed-order reduction => deterministic (no atomics).
__global__ void __launch_bounds__(256)
k_sample_pred(const float* __restrict__ masks, const Tap* __restrict__ taps,
              _Float16* __restrict__ A1, _Float16* __restrict__ A2,
              float* __restrict__ neg_sum, float* __restrict__ sig_sum) {
    int r = blockIdx.x;                          // 0..399
    const float* base = masks + (size_t)r * HW;
    float accN = 0.f, accS = 0.f;
    for (int p = threadIdx.x; p < NPTS; p += 256) {
        Tap t = taps[p];
        float os = base[t.o[0]]*t.w[0] + base[t.o[1]]*t.w[1]
                 + base[t.o[2]]*t.w[2] + base[t.o[3]]*t.w[3];
        float sig = 1.f / (1.f + expf(-os));
        float sp  = fmaxf(os, 0.f) + log1pf(expf(-fabsf(os)));  // softplus(os)
        accN += sp;
        accS += sig;
        A1[(size_t)r*NPTS + p] = (_Float16)(-os);
        A2[(size_t)r*NPTS + p] = (_Float16)sig;
    }
    __shared__ float sN[256], sS[256];
    sN[threadIdx.x] = accN; sS[threadIdx.x] = accS;
    __syncthreads();
    for (int d = 128; d > 0; d >>= 1) {
        if (threadIdx.x < d) {
            sN[threadIdx.x] += sN[threadIdx.x + d];
            sS[threadIdx.x] += sS[threadIdx.x + d];
        }
        __syncthreads();
    }
    if (threadIdx.x == 0) { neg_sum[r] = sN[0]; sig_sum[r] = sS[0]; }
}

// ---- Kernel 3: sample target masks -> Bt (f16) + colsum; rows 200..207 = 0 -
__global__ void __launch_bounds__(256)
k_sample_tgt(const float* __restrict__ masks, const Tap* __restrict__ taps,
             _Float16* __restrict__ Bt, float* __restrict__ ts_sum) {
    int r = blockIdx.x;                          // 0..207
    float acc = 0.f;
    if (r < NCOL) {
        const float* base = masks + (size_t)r * HW;
        for (int p = threadIdx.x; p < NPTS; p += 256) {
            Tap t = taps[p];
            float ts = base[t.o[0]]*t.w[0] + base[t.o[1]]*t.w[1]
                     + base[t.o[2]]*t.w[2] + base[t.o[3]]*t.w[3];
            acc += ts;
            Bt[(size_t)r*NPTS + p] = (_Float16)ts;
        }
    } else {
        for (int p = threadIdx.x; p < NPTS; p += 256)
            Bt[(size_t)r*NPTS + p] = (_Float16)0.f;   // zero pad rows for GEMM
    }
    __shared__ float s[256];
    s[threadIdx.x] = acc;
    __syncthreads();
    for (int d = 128; d > 0; d >>= 1) {
        if (threadIdx.x < d) s[threadIdx.x] += s[threadIdx.x + d];
        __syncthreads();
    }
    if (threadIdx.x == 0) ts_sum[r] = s[0];
}

// ---------------- Kernel 4: softmax over 81 classes, one wave32 per row -----
__global__ void __launch_bounds__(256)
k_softmax(const float* __restrict__ logits, float* __restrict__ prob) {
    int wave = threadIdx.x >> 5;
    int lane = threadIdx.x & 31;
    int r = blockIdx.x * 8 + wave;
    if (r >= NROW) return;
    const float* row = logits + (size_t)r * NCLS;
    float m = -1e30f;
    for (int j = lane; j < NCLS; j += 32) m = fmaxf(m, row[j]);
    for (int d = 16; d > 0; d >>= 1) m = fmaxf(m, __shfl_xor(m, d, 32));
    float s = 0.f;
    for (int j = lane; j < NCLS; j += 32) s += expf(row[j] - m);
    for (int d = 16; d > 0; d >>= 1) s += __shfl_xor(s, d, 32);
    float inv = 1.f / s;
    for (int j = lane; j < NCLS; j += 32)
        prob[(size_t)r*NCLS + j] = expf(row[j] - m) * inv;
}

// ------- Kernel 5: dual WMMA GEMM (shared B operand) + fused epilogue -------
// One 16x16 output tile per wave; 8 waves/block; 25*13 = 325 tiles.
// A/B fragments: lane holds (row M%16 / col N%16), K chunks [k0..k0+7] and
// [k0+16..k0+23] with k0 = (lane<16 ? 0 : 8)  -> two aligned b128 loads each.
__global__ void __launch_bounds__(256)
k_gemm_epilogue(const _Float16* __restrict__ A1, const _Float16* __restrict__ A2,
                const _Float16* __restrict__ Bt,
                const float* __restrict__ neg_sum, const float* __restrict__ sig_sum,
                const float* __restrict__ ts_sum,  const float* __restrict__ prob,
                const int* __restrict__ tgt_ids,   float* __restrict__ out) {
    int tid  = threadIdx.x;
    int wave = tid >> 5;
    int lane = tid & 31;
    int tile = blockIdx.x * 8 + wave;
    if (tile >= MT * NT) return;                 // wave-uniform skip
    int mt = tile / NT, nt = tile % NT;
    int rl = lane & 15;
    int hi = lane >> 4;                          // 0: K base 0, 1: K base 8

    const _Float16* a1p = A1 + ((size_t)(mt*16 + rl)) * NPTS + hi*8;
    const _Float16* a2p = A2 + ((size_t)(mt*16 + rl)) * NPTS + hi*8;
    const _Float16* bp  = Bt + ((size_t)(nt*16 + rl)) * NPTS + hi*8;

    v8f c1 = {};   // (-os) @ ts^T
    v8f c2 = {};   // sig(os) @ ts^T

#pragma unroll 4
    for (int k = 0; k < NPTS; k += 32) {
        v8h a1lo = *(const v8h*)(a1p + k);
        v8h a1hi = *(const v8h*)(a1p + k + 16);
        v8h a2lo = *(const v8h*)(a2p + k);
        v8h a2hi = *(const v8h*)(a2p + k + 16);
        v8h blo  = *(const v8h*)(bp  + k);
        v8h bhi  = *(const v8h*)(bp  + k + 16);
        v16h af1 = __builtin_shufflevector(a1lo, a1hi,
                      0,1,2,3,4,5,6,7,8,9,10,11,12,13,14,15);
        v16h af2 = __builtin_shufflevector(a2lo, a2hi,
                      0,1,2,3,4,5,6,7,8,9,10,11,12,13,14,15);
        v16h bf  = __builtin_shufflevector(blo,  bhi,
                      0,1,2,3,4,5,6,7,8,9,10,11,12,13,14,15);
        c1 = __builtin_amdgcn_wmma_f32_16x16x32_f16(false, af1, false, bf,
                                                    (short)0, c1, false, false);
        c2 = __builtin_amdgcn_wmma_f32_16x16x32_f16(false, af2, false, bf,
                                                    (short)0, c2, false, false);
    }

    int col = nt*16 + rl;
    if (col >= NCOL) return;                     // pad columns (after WMMAs)
    float tss  = ts_sum[col];
    int   tcls = tgt_ids[col];
    const float invP = 1.f / (float)NPTS;

#pragma unroll
    for (int v = 0; v < 8; v++) {
        int row = mt*16 + v + hi*8;              // D layout: M = v + 8*(lane>=16)
        float cmask = (c1[v] + neg_sum[row]) * invP;
        float denom = sig_sum[row] + tss + 1.f;
        float cdice = 1.f - (2.f*c2[v] + 1.f) / denom;
        float ccls  = -prob[(size_t)row*NCLS + tcls];
        float o = 5.f*cmask + 2.f*ccls + 5.f*cdice;
        if (__builtin_isnan(o))      o = 1e8f;
        else if (__builtin_isinf(o)) o = (o > 0.f) ? 1e8f : -1e8f;
        out[(size_t)row*NCOL + col] = o;
    }
}

// ---------------------------------------------------------------------------
extern "C" void kernel_launch(void* const* d_in, const int* in_sizes, int n_in,
                              void* d_out, int out_size, void* d_ws, size_t ws_size,
                              hipStream_t stream) {
    const float* pred_logits = (const float*)d_in[0];   // [4,100,81]
    const float* pred_masks  = (const float*)d_in[1];   // [4,100,256,256]
    const float* tgt_masks   = (const float*)d_in[2];   // [4,50,256,256]
    const float* coords      = (const float*)d_in[3];   // [1,12544,2]
    const int*   tgt_labels  = (const int*)d_in[4];     // [200]
    float* out = (float*)d_out;                         // [4,100,200]

    // Workspace carve-out (~25.9 MB total), 256B-aligned regions.
    char* w = (char*)d_ws;
    size_t off = 0;
    auto alignup = [](size_t x) { return (x + 255) & ~(size_t)255; };

    Tap*      taps    = (Tap*)(w + off);      off = alignup(off + sizeof(Tap) * NPTS);
    _Float16* A1      = (_Float16*)(w + off); off = alignup(off + (size_t)2 * NROW  * NPTS);
    _Float16* A2      = (_Float16*)(w + off); off = alignup(off + (size_t)2 * NROW  * NPTS);
    _Float16* Bt      = (_Float16*)(w + off); off = alignup(off + (size_t)2 * NCOLP * NPTS);
    float*    neg_sum = (float*)(w + off);    off = alignup(off + 4 * NROW);
    float*    sig_sum = (float*)(w + off);    off = alignup(off + 4 * NROW);
    float*    ts_sum  = (float*)(w + off);    off = alignup(off + 4 * NCOLP);
    float*    prob    = (float*)(w + off);    off = alignup(off + 4 * (size_t)NROW * NCLS);
    (void)ws_size; (void)in_sizes; (void)n_in; (void)out_size;

    k_taps<<<(NPTS + 255) / 256, 256, 0, stream>>>(coords, taps);
    k_sample_pred<<<NROW, 256, 0, stream>>>(pred_masks, taps, A1, A2,
                                            neg_sum, sig_sum);
    k_sample_tgt<<<NCOLP, 256, 0, stream>>>(tgt_masks, taps, Bt, ts_sum);
    k_softmax<<<NROW / 8, 256, 0, stream>>>(pred_logits, prob);
    k_gemm_epilogue<<<(MT * NT + 7) / 8, 256, 0, stream>>>(
        A1, A2, Bt, neg_sum, sig_sum, ts_sum, prob, tgt_labels, out);
}
